// SerializedAttention_27874337751667
// MI455X (gfx1250) — compile-verified, matrix operand order checked by reading, not run on hardware
//
#include <hip/hip_runtime.h>
#include <cstdint>

// ---------------------------------------------------------------------------
// Serialized windowed attention for MI455X (gfx1250, wave32, WMMA + TDM).
//
// Pipeline (f32 accumulate, f16 matrix inputs -> v_wmma_f32_16x16x32_f16):
//   1) qkv  = gather(feat, order) @ Wqkv + bqkv      -> f16 ws   [N,1536]
//   2) per (window, head): S = Q K^T, softmax(fp32), O = P V
//      Q/K tiles staged global->LDS by the Tensor Data Mover (TDM) with
//      hardware row padding (stride 64 -> 72 halves); scattered by order
//                                                     -> f16 ws   [N,512]
//   3) out  = attn @ Wproj + bproj                    -> f32 d_out
//
// Workspace use: N*1536*2 + N*512*2 = 512 MB (assumed <= ws_size).
// ---------------------------------------------------------------------------

#define N_PTS   131072
#define C_DIM   512
#define K_PATCH 128
#define H_HEADS 8

typedef __attribute__((ext_vector_type(16))) _Float16 v16h;
typedef __attribute__((ext_vector_type(8)))  _Float16 v8h;
typedef __attribute__((ext_vector_type(8)))  float    v8f;
typedef __attribute__((ext_vector_type(4)))  unsigned int u32x4;
typedef __attribute__((ext_vector_type(8)))  unsigned int u32x8;

static __device__ __forceinline__ v8f wmma_f16(v16h a, v16h b, v8f c) {
  // 8 args: (neg_a, A, neg_b, B, c_mod, C, reuse_a, reuse_b)
  return __builtin_amdgcn_wmma_f32_16x16x32_f16(false, a, false, b, (short)0, c,
                                                false, false);
}

// Assemble one 16-half fragment from two 16-byte LDS chunks (ds_load_b128 x2).
// Per ISA layout (16-bit A/B 16x32): elems 0-7 at K = 8*(lane/16)+[0..7],
// elems 8-15 at K = 16 + 8*(lane/16)+[0..7]  (relative to the 32-wide K block).
static __device__ __forceinline__ v16h lds_frag(const _Float16* p) {
  v8h lo = *(const v8h*)(p);
  v8h hi = *(const v8h*)(p + 16);
  return __builtin_shufflevector(lo, hi, 0, 1, 2, 3, 4, 5, 6, 7,
                                 8, 9, 10, 11, 12, 13, 14, 15);
}

static __device__ __forceinline__ v8f vzero8() {
  v8f z = {0.f, 0.f, 0.f, 0.f, 0.f, 0.f, 0.f, 0.f};
  return z;
}

// TDM: DMA a 2D tile of 128 rows x 64 f16 from a row-major [*,1536] f16 tensor
// into LDS, inserting 16 B of pad after every 128 B row (LDS stride 72 halves).
// Descriptor per CDNA5 ISA ch.8 (D# groups 0/1; groups 2/3 omitted -> 2D).
static __device__ __forceinline__ void tdm_load_tile_128x64(const _Float16* gsrc,
                                                            unsigned int ldsByteOff) {
  const unsigned long long ga = (unsigned long long)(uintptr_t)gsrc;
  u32x4 g0;
  g0[0] = 1u;                                           // count=1 (valid user D#)
  g0[1] = ldsByteOff;                                   // lds_addr (bytes)
  g0[2] = (unsigned int)ga;                             // global_addr[31:0]
  g0[3] = (unsigned int)((ga >> 32) & 0x01FFFFFFu)      // global_addr[56:32]
          | (2u << 30);                                 // type=2 ("image")
  u32x8 g1;
  g1[0] = (1u << 16)      // data_size = 1 -> 2-byte elements
        | (1u << 20)      // pad_enable
        | (4u << 22)      // pad_interval: 2^(4+1)=32 DWORDs (= one 64-half row)
        | (3u << 25);     // pad_amount: 3+1 = 4 DWORDs (= 16 B)
  g1[1] = (1536u << 16);          // tensor_dim0[15:0] (bits 63:48)
  g1[2] = 0xFFFF0000u;            // tensor_dim0[31:16]=0 | tensor_dim1[15:0]
  g1[3] = 0x7FFFu | (64u << 16);  // tensor_dim1[31:16] | tile_dim0 = 64
  g1[4] = 128u;                   // tile_dim1 = 128 rows (tile_dim2 = 0)
  g1[5] = 1536u;                  // tensor_dim0_stride[31:0] (elements)
  g1[6] = 0u;                     // stride[47:32] | tensor_dim1_stride[15:0]
  g1[7] = 0u;
  asm volatile("tensor_load_to_lds %0, %1" :: "s"(g0), "s"(g1) : "memory");
}

// ---------------------------------------------------------------------------
// Kernel 1: gather-fused QKV GEMM.  Grid (1024, 12), 256 threads (8 waves).
// Each block: 128x128 tile of qkv[j,:] = feat[order[j],:] @ Wqkv + bqkv.
// ---------------------------------------------------------------------------
__global__ __launch_bounds__(256) void
qkv_gemm_kernel(const float* __restrict__ feat, const float* __restrict__ Wqkv,
                const float* __restrict__ bqkv, const long long* __restrict__ order,
                _Float16* __restrict__ qkvOut)
{
  __shared__ _Float16 Alds[128 * 72];   // [row][k], stride 72 -> conflict-free b128
  __shared__ _Float16 Btlds[128 * 72];  // [col][k] (B transposed)
  __shared__ int rowIdx[128];

  const int tid = threadIdx.x;
  const int wid = tid >> 5;
  const int lane = tid & 31;
  const int lm = lane & 15;             // tile row (A) / tile col (B)
  const int lchunk = (lane >> 4) * 8;   // K chunk select per half-wave

  const int rowBase = blockIdx.x * 128;
  const int colBase = blockIdx.y * 128;

  if (tid < 128) rowIdx[tid] = (int)order[rowBase + tid];
  __syncthreads();

  v8f acc[8];
#pragma unroll
  for (int i = 0; i < 8; ++i) acc[i] = vzero8();

  const int ar = tid >> 1;            // A: 2 threads per row, 32 floats each
  const int ak = (tid & 1) * 32;
  const int bk = tid >> 2;            // B: 4 threads per k-row, 32 floats each
  const int bn = (tid & 3) * 32;

  for (int kb = 0; kb < C_DIM; kb += 64) {
    {  // stage A (gathered, fp32 -> f16)
      const float* src = feat + (long)rowIdx[ar] * C_DIM + kb + ak;
      _Float16* dst = Alds + ar * 72 + ak;
#pragma unroll
      for (int i = 0; i < 8; ++i) {
        float4 f = ((const float4*)src)[i];
        dst[4 * i + 0] = (_Float16)f.x; dst[4 * i + 1] = (_Float16)f.y;
        dst[4 * i + 2] = (_Float16)f.z; dst[4 * i + 3] = (_Float16)f.w;
      }
    }
    {  // stage B transposed (fp32 -> f16)
      const float* src = Wqkv + (long)(kb + bk) * 1536 + colBase + bn;
#pragma unroll
      for (int i = 0; i < 8; ++i) {
        float4 f = ((const float4*)src)[i];
        Btlds[(bn + 4 * i + 0) * 72 + bk] = (_Float16)f.x;
        Btlds[(bn + 4 * i + 1) * 72 + bk] = (_Float16)f.y;
        Btlds[(bn + 4 * i + 2) * 72 + bk] = (_Float16)f.z;
        Btlds[(bn + 4 * i + 3) * 72 + bk] = (_Float16)f.w;
      }
    }
    // prefetch next stage while this stage's WMMAs run (global_prefetch_b8)
    if (kb + 64 < C_DIM) {
      __builtin_prefetch(feat + (long)rowIdx[ar] * C_DIM + kb + 64 + ak, 0, 0);
      __builtin_prefetch(Wqkv + (long)(kb + 64 + bk) * 1536 + colBase + bn, 0, 0);
    }
    __syncthreads();

    const _Float16* arow = Alds + (wid * 16 + lm) * 72;
    v16h a0 = lds_frag(arow + lchunk);
    v16h a1 = lds_frag(arow + 32 + lchunk);
#pragma unroll
    for (int nt = 0; nt < 8; ++nt) {
      const _Float16* brow = Btlds + (nt * 16 + lm) * 72;
      acc[nt] = wmma_f16(a0, lds_frag(brow + lchunk), acc[nt]);
      acc[nt] = wmma_f16(a1, lds_frag(brow + 32 + lchunk), acc[nt]);
    }
    __syncthreads();
  }

  // epilogue: C layout lane L VGPR j -> (M = 8*(L/16)+j, N = L%16)
#pragma unroll
  for (int nt = 0; nt < 8; ++nt) {
    const int col = colBase + nt * 16 + lm;
    const float bv = bqkv[col];
#pragma unroll
    for (int j = 0; j < 8; ++j) {
      const int row = rowBase + wid * 16 + (lane >> 4) * 8 + j;
      qkvOut[(long)row * 1536 + col] = (_Float16)(acc[nt][j] + bv);
    }
  }
}

// ---------------------------------------------------------------------------
// Kernel 2: windowed attention.  Grid (1024 windows, 8 heads), 256 threads.
// Dynamic LDS: Q[128x72] K[128x72] Vt[64x136] (f16), S[128x129] (f32),
//              P[128x136] (f16), rowIdx[128] (int)  = 155,648 B.
// Q and K tiles are loaded by the Tensor Data Mover with row padding.
// ---------------------------------------------------------------------------
#define ATTN_SMEM_BYTES 155648
#define Q_LDS_OFF   0
#define K_LDS_OFF   18432

__global__ __launch_bounds__(256) void
attn_kernel(const _Float16* __restrict__ qkv, const long long* __restrict__ order,
            _Float16* __restrict__ attnOut)
{
  extern __shared__ char smem[];
  _Float16* Qs  = (_Float16*)(smem + Q_LDS_OFF);   // 128*72*2 = 18432
  _Float16* Ks  = (_Float16*)(smem + K_LDS_OFF);   // 18432
  _Float16* Vts = (_Float16*)(smem + 36864);       // 64*136*2 = 17408
  float*    Ss  = (float*)   (smem + 54272);       // 128*129*4 = 66048
  _Float16* Ps  = (_Float16*)(smem + 120320);      // 128*136*2 = 34816
  int*      rIdx = (int*)    (smem + 155136);      // 512

  const int tid = threadIdx.x;
  const int wid = tid >> 5;
  const int lane = tid & 31;
  const int lm = lane & 15;
  const int lchunk = (lane >> 4) * 8;
  const int head = blockIdx.y;
  const long gBase = (long)blockIdx.x * K_PATCH;

  // TDM DMA of Q and K tiles (one issue per workgroup; TENSORcnt on wave 0)
  if (tid == 0) {
    tdm_load_tile_128x64(qkv + gBase * 1536 + head * 64, Q_LDS_OFF);
    tdm_load_tile_128x64(qkv + gBase * 1536 + 512 + head * 64, K_LDS_OFF);
  }

  if (tid < 128) rIdx[tid] = (int)order[gBase + tid];

  {  // V loaded manually (TDM cannot transpose): Vt[d][t]
    const int t = tid >> 1;
    const int dh = (tid & 1) * 32;
    const _Float16* base = qkv + (gBase + t) * 1536 + head * 64 + dh;
#pragma unroll
    for (int i = 0; i < 4; ++i) {
      v8h vv = *(const v8h*)(base + 1024 + 8 * i);
#pragma unroll
      for (int e = 0; e < 8; ++e) Vts[(dh + 8 * i + e) * 136 + t] = vv[e];
    }
  }
  __builtin_amdgcn_s_wait_tensorcnt(0);   // wave 0 drains its TDM ops
  __syncthreads();                        // publishes Q/K/V to all waves

  {  // S strip (16 rows x 128 cols) per wave:  S = Q K^T  (f32 acc)
    const _Float16* qrow = Qs + (wid * 16 + lm) * 72;
    v16h a0 = lds_frag(qrow + lchunk);
    v16h a1 = lds_frag(qrow + 32 + lchunk);
#pragma unroll
    for (int nt = 0; nt < 8; ++nt) {
      const _Float16* krow = Ks + (nt * 16 + lm) * 72;
      v8f s = vzero8();
      s = wmma_f16(a0, lds_frag(krow + lchunk), s);
      s = wmma_f16(a1, lds_frag(krow + 32 + lchunk), s);
#pragma unroll
      for (int j = 0; j < 8; ++j)
        Ss[(wid * 16 + (lane >> 4) * 8 + j) * 129 + nt * 16 + lm] = s[j];
    }
  }
  __syncthreads();

  // row softmax in fp32 (scale = hd^-0.5 = 0.125 folded into exp argument)
  if (tid < 128) {
    float* srow = Ss + tid * 129;
    float m = -3.4e38f;
    for (int c = 0; c < 128; ++c) m = fmaxf(m, srow[c]);
    float sum = 0.f;
    for (int c = 0; c < 128; ++c) {
      float e = __expf((srow[c] - m) * 0.125f);
      srow[c] = e;
      sum += e;
    }
    const float inv = 1.f / sum;
    _Float16* prow = Ps + tid * 136;
    for (int c = 0; c < 128; ++c) prow[c] = (_Float16)(srow[c] * inv);
  }
  __syncthreads();

  // O strip (16 rows x 64 cols) per wave:  O = P V   (K = 128 -> 4 steps)
  v8f acc[4];
#pragma unroll
  for (int i = 0; i < 4; ++i) acc[i] = vzero8();
  const _Float16* prow = Ps + (wid * 16 + lm) * 136;
#pragma unroll
  for (int kbt = 0; kbt < 4; ++kbt) {
    v16h a = lds_frag(prow + kbt * 32 + lchunk);
#pragma unroll
    for (int nt = 0; nt < 4; ++nt) {
      const _Float16* vrow = Vts + (nt * 16 + lm) * 136;
      acc[nt] = wmma_f16(a, lds_frag(vrow + kbt * 32 + lchunk), acc[nt]);
    }
  }

  // scatter store by order (makes the proj GEMM read contiguous; `inverse`
  // is not needed since order[inverse[i]] == i)
  int prows[8];
#pragma unroll
  for (int j = 0; j < 8; ++j) prows[j] = rIdx[wid * 16 + (lane >> 4) * 8 + j];
#pragma unroll
  for (int nt = 0; nt < 4; ++nt) {
    const int col = head * 64 + nt * 16 + lm;
#pragma unroll
    for (int j = 0; j < 8; ++j)
      attnOut[(long)prows[j] * 512 + col] = (_Float16)acc[nt][j];
  }
}

// ---------------------------------------------------------------------------
// Kernel 3: output projection GEMM.  Grid (1024, 4), 256 threads.
// out = attn(f16) @ Wproj + bproj  -> f32
// ---------------------------------------------------------------------------
__global__ __launch_bounds__(256) void
proj_gemm_kernel(const _Float16* __restrict__ attnIn, const float* __restrict__ Wproj,
                 const float* __restrict__ bproj, float* __restrict__ out)
{
  __shared__ _Float16 Alds[128 * 72];
  __shared__ _Float16 Btlds[128 * 72];

  const int tid = threadIdx.x;
  const int wid = tid >> 5;
  const int lane = tid & 31;
  const int lm = lane & 15;
  const int lchunk = (lane >> 4) * 8;

  const int rowBase = blockIdx.x * 128;
  const int colBase = blockIdx.y * 128;

  v8f acc[8];
#pragma unroll
  for (int i = 0; i < 8; ++i) acc[i] = vzero8();

  const int ar = tid >> 1;
  const int ak = (tid & 1) * 32;
  const int bk = tid >> 2;
  const int bn = (tid & 3) * 32;

  for (int kb = 0; kb < C_DIM; kb += 64) {
    {  // stage A (already f16): 4x b128 copies
      const _Float16* src = attnIn + (long)(rowBase + ar) * 512 + kb + ak;
      _Float16* dst = Alds + ar * 72 + ak;
#pragma unroll
      for (int i = 0; i < 4; ++i)
        *(v8h*)(dst + 8 * i) = *(const v8h*)(src + 8 * i);
    }
    {  // stage B transposed (fp32 -> f16)
      const float* src = Wproj + (long)(kb + bk) * 512 + colBase + bn;
#pragma unroll
      for (int i = 0; i < 8; ++i) {
        float4 f = ((const float4*)src)[i];
        Btlds[(bn + 4 * i + 0) * 72 + bk] = (_Float16)f.x;
        Btlds[(bn + 4 * i + 1) * 72 + bk] = (_Float16)f.y;
        Btlds[(bn + 4 * i + 2) * 72 + bk] = (_Float16)f.z;
        Btlds[(bn + 4 * i + 3) * 72 + bk] = (_Float16)f.w;
      }
    }
    if (kb + 64 < C_DIM) {
      __builtin_prefetch(attnIn + (long)(rowBase + ar) * 512 + kb + 64 + ak, 0, 0);
      __builtin_prefetch(Wproj + (long)(kb + 64 + bk) * 512 + colBase + bn, 0, 0);
    }
    __syncthreads();

    const _Float16* arow = Alds + (wid * 16 + lm) * 72;
    v16h a0 = lds_frag(arow + lchunk);
    v16h a1 = lds_frag(arow + 32 + lchunk);
#pragma unroll
    for (int nt = 0; nt < 8; ++nt) {
      const _Float16* brow = Btlds + (nt * 16 + lm) * 72;
      acc[nt] = wmma_f16(a0, lds_frag(brow + lchunk), acc[nt]);
      acc[nt] = wmma_f16(a1, lds_frag(brow + 32 + lchunk), acc[nt]);
    }
    __syncthreads();
  }

#pragma unroll
  for (int nt = 0; nt < 8; ++nt) {
    const int col = colBase + nt * 16 + lm;
    const float bv = bproj[col];
#pragma unroll
    for (int j = 0; j < 8; ++j) {
      const int row = rowBase + wid * 16 + (lane >> 4) * 8 + j;
      out[(long)row * 512 + col] = acc[nt][j] + bv;
    }
  }
}

// ---------------------------------------------------------------------------
extern "C" void kernel_launch(void* const* d_in, const int* in_sizes, int n_in,
                              void* d_out, int out_size, void* d_ws, size_t ws_size,
                              hipStream_t stream) {
  (void)in_sizes; (void)n_in; (void)out_size; (void)ws_size;

  const float* feat      = (const float*)d_in[0];
  const float* Wqkv      = (const float*)d_in[1];
  const float* bqkv      = (const float*)d_in[2];
  const float* Wproj     = (const float*)d_in[3];
  const float* bproj     = (const float*)d_in[4];
  const long long* order = (const long long*)d_in[5];
  // d_in[6] = inverse: unused (attention output is scattered by `order`).
  float* out = (float*)d_out;

  _Float16* qkvWs  = (_Float16*)d_ws;                               // N*1536 f16
  _Float16* attnWs = (_Float16*)((char*)d_ws +
                                 (size_t)N_PTS * 1536 * sizeof(_Float16)); // N*512 f16

  qkv_gemm_kernel<<<dim3(N_PTS / 128, 1536 / 128), 256, 0, stream>>>(
      feat, Wqkv, bqkv, order, qkvWs);
  attn_kernel<<<dim3(N_PTS / K_PATCH, H_HEADS), 256, ATTN_SMEM_BYTES, stream>>>(
      qkvWs, order, attnWs);
  proj_gemm_kernel<<<dim3(N_PTS / 128, C_DIM / 128), 256, 0, stream>>>(
      attnWs, Wproj, bproj, out);
}
